// Llama4TextMoe_1614907703548
// MI455X (gfx1250) — matrix operand order repaired and checked by reading, not compiled, not run on hardware
//
#include <hip/hip_runtime.h>
#include <hip/hip_bf16.h>

#define T_TOK 4096
#define DIM   1024
#define FDIM  2048
#define NEXP  8
#define BM 128
#define BN 128
#define BK 32
#define LDK 40                 // padded LDS K-stride (bf16 elems), 80B rows keep 16B alignment
#define MT_PER_E (T_TOK / BM)  // 32

typedef __attribute__((ext_vector_type(16))) __bf16 v16bf;
typedef __attribute__((ext_vector_type(8)))  __bf16 v8bf;
typedef __attribute__((ext_vector_type(8)))  float  v8f;
typedef __attribute__((ext_vector_type(8)))  unsigned short us8;
typedef __attribute__((ext_vector_type(4)))  unsigned short us4;
typedef __attribute__((ext_vector_type(4)))  int v4i;

#define HAS_ASYNC_LDS __has_builtin(__builtin_amdgcn_global_load_async_to_lds_b128)

static __device__ __forceinline__ unsigned short f2bf(float f) {
  unsigned int u = __builtin_bit_cast(unsigned int, f);
  u += 0x7FFFu + ((u >> 16) & 1u);          // round-to-nearest-even
  return (unsigned short)(u >> 16);
}
static __device__ __forceinline__ float bf2f(unsigned short h) {
  return __builtin_bit_cast(float, ((unsigned int)h) << 16);
}
// g * sigmoid(g) with fast v_rcp_f32 (bandwidth-bound kernel; ~1ulp is fine)
static __device__ __forceinline__ float fast_silu(float g) {
  return g * __builtin_amdgcn_rcpf(1.f + __expf(-g));
}

// ---- CDNA5 async global->LDS (ASYNCcnt DMA), guarded so compile stays green
// Builtin signature (from hipcc diagnostic): param0 = AS1 pointer to int4.
static __device__ __forceinline__ void async_ld_b128(const void* gp, void* lp) {
#if HAS_ASYNC_LDS
  __builtin_amdgcn_global_load_async_to_lds_b128(
      (__attribute__((address_space(1))) v4i*)(unsigned long long)gp,
      (__attribute__((address_space(3))) v4i*)(unsigned int)(unsigned long long)lp,
      0, 0);
#else
  *(us8*)lp = *(const us8*)gp;
#endif
}
static __device__ __forceinline__ void wait_async() {
#if HAS_ASYNC_LDS
#if __has_builtin(__builtin_amdgcn_s_wait_asynccnt)
  __builtin_amdgcn_s_wait_asynccnt(0);
#else
  asm volatile("s_wait_asynccnt 0x0" ::: "memory");
#endif
#endif
}

// A fragment (16x32 bf16, [M][K] row-major in LDS):
// lanes 0-15: M=lane, elems 0..7 = K 0..7, elems 8..15 = K 16..23
// lanes 16-31: M=lane-16, elems 0..7 = K 8..15, elems 8..15 = K 24..31
static __device__ __forceinline__ v16bf frag_ld_A(const unsigned short* base, int lane) {
  const int m = lane & 15, hh = lane >> 4;
  const unsigned short* p = base + m * LDK;
  us8 a = *(const us8*)(p + hh * 8);
  us8 b = *(const us8*)(p + 16 + hh * 8);
  v8bf x = __builtin_bit_cast(v8bf, a);
  v8bf y = __builtin_bit_cast(v8bf, b);
  return __builtin_shufflevector(x, y, 0,1,2,3,4,5,6,7,8,9,10,11,12,13,14,15);
}
// B fragment (32x16 bf16, stored [N][K] in LDS):
// lanes 0-15: N=lane, elems 0..7 = K 0..7, elems 8..15 = K 16..23 ; lanes 16-31: N=lane-16, K 8..15, 24..31
static __device__ __forceinline__ v16bf frag_ld_B(const unsigned short* base, int lane) {
  const int n = lane & 15, hh = lane >> 4;
  const unsigned short* p = base + n * LDK + hh * 8;
  us8 a = *(const us8*)(p);
  us8 b = *(const us8*)(p + 16);
  v8bf x = __builtin_bit_cast(v8bf, a);
  v8bf y = __builtin_bit_cast(v8bf, b);
  return __builtin_shufflevector(x, y, 0,1,2,3,4,5,6,7,8,9,10,11,12,13,14,15);
}
#define WMMA_BF16(A,B,C) \
  __builtin_amdgcn_wmma_f32_16x16x32_bf16(false,(A),false,(B),(short)0,(C),false,false)

// ---------------------------------------------------------------- utility

__global__ __launch_bounds__(256) void k_convert_x(const float* __restrict__ x,
                                                   unsigned short* __restrict__ xb) {
  size_t i = ((size_t)blockIdx.x * 256 + threadIdx.x) * 4;
  float4 v = *(const float4*)(x + i);
  us4 o; o.x = f2bf(v.x); o.y = f2bf(v.y); o.z = f2bf(v.z); o.w = f2bf(v.w);
  *(us4*)(xb + i) = o;
}

__global__ __launch_bounds__(256) void k_router(const float* __restrict__ x,
                                                const float* __restrict__ rw,
                                                int* __restrict__ tope,
                                                float* __restrict__ scl,
                                                int* __restrict__ cnt) {
  const int wid = threadIdx.x >> 5, lane = threadIdx.x & 31;
  const int t = blockIdx.x * 8 + wid;
  const float* xr = x + (size_t)t * DIM;
  float acc[NEXP] = {};
  for (int d = lane; d < DIM; d += 32) {
    float xv = xr[d];
#pragma unroll
    for (int e = 0; e < NEXP; ++e) acc[e] += xv * rw[e * DIM + d];
  }
#pragma unroll
  for (int e = 0; e < NEXP; ++e)
#pragma unroll
    for (int off = 16; off; off >>= 1) acc[e] += __shfl_down(acc[e], off, 32);
  if (lane == 0) {
    float best = acc[0]; int be = 0;
#pragma unroll
    for (int e = 1; e < NEXP; ++e) if (acc[e] > best) { best = acc[e]; be = e; }
    tope[t] = be;
    scl[t]  = __builtin_amdgcn_rcpf(1.f + __expf(-best));
    atomicAdd(&cnt[be], 1);
  }
}

__global__ void k_scan8(const int* __restrict__ cnt, int* __restrict__ offs,
                        int* __restrict__ curs) {
  if (threadIdx.x == 0) {
    int s = 0;
    for (int e = 0; e < NEXP; ++e) { offs[e] = s; curs[e] = s; s += cnt[e]; }
  }
}

__global__ __launch_bounds__(256) void k_assign(const int* __restrict__ tope,
                                                int* __restrict__ curs,
                                                int* __restrict__ perm) {
  int t = blockIdx.x * 256 + threadIdx.x;
  int e = tope[t];
  int p = atomicAdd(&curs[e], 1);
  perm[p] = t;
}

// ---------------------------------------------------------------- shared path

__global__ __launch_bounds__(256) void k_shared_gateup(
    const unsigned short* __restrict__ xb,   // [T][D] bf16
    const float* __restrict__ wg,            // [F][D]  (already [N][K])
    const float* __restrict__ wu,            // [F][D]
    unsigned short* __restrict__ h) {        // [T][F] bf16
  __shared__ __align__(16) unsigned short lA[BM * LDK];
  __shared__ __align__(16) unsigned short lG[BN * LDK];
  __shared__ __align__(16) unsigned short lU[BN * LDK];
  const int m0 = blockIdx.x * BM, n0 = blockIdx.y * BN;
  const int tid = threadIdx.x, lane = tid & 31, wid = tid >> 5;
  const int wm = (wid >> 2) * 64, wn = (wid & 3) * 32;
  v8f cg[4][2] = {}; v8f cu[4][2] = {};
  for (int k0 = 0; k0 < DIM; k0 += BK) {
    __syncthreads();
    // A tile: pure byte-copy -> async DMA to LDS (overlaps with B conversion below)
    for (int task = tid; task < BM * 4; task += 256) {
      int r = task >> 2, v = (task & 3) * 8;
      async_ld_b128(xb + (size_t)(m0 + r) * DIM + k0 + v, lA + r * LDK + v);
    }
    for (int task = tid; task < BN * 8; task += 256) {
      int r = task >> 3, v = (task & 7) * 4;
      const float* gp = wg + (size_t)(n0 + r) * DIM + k0 + v;
      const float* up = wu + (size_t)(n0 + r) * DIM + k0 + v;
      if ((task & 7) == 0 && k0 + BK < DIM) {
        __builtin_prefetch(gp + BK, 0, 0);
        __builtin_prefetch(up + BK, 0, 0);
      }
      float4 g4 = *(const float4*)gp;
      us4 gg; gg.x = f2bf(g4.x); gg.y = f2bf(g4.y); gg.z = f2bf(g4.z); gg.w = f2bf(g4.w);
      *(us4*)(lG + r * LDK + v) = gg;
      float4 u4 = *(const float4*)up;
      us4 uu; uu.x = f2bf(u4.x); uu.y = f2bf(u4.y); uu.z = f2bf(u4.z); uu.w = f2bf(u4.w);
      *(us4*)(lU + r * LDK + v) = uu;
    }
    wait_async();
    __syncthreads();
    v16bf a[4];
#pragma unroll
    for (int i = 0; i < 4; ++i) a[i] = frag_ld_A(lA + (wm + i * 16) * LDK, lane);
#pragma unroll
    for (int j = 0; j < 2; ++j) {
      v16bf bg = frag_ld_B(lG + (wn + j * 16) * LDK, lane);
      v16bf bu = frag_ld_B(lU + (wn + j * 16) * LDK, lane);
#pragma unroll
      for (int i = 0; i < 4; ++i) {
        cg[i][j] = WMMA_BF16(a[i], bg, cg[i][j]);
        cu[i][j] = WMMA_BF16(a[i], bu, cu[i][j]);
      }
    }
  }
  const int nlane = lane & 15, half = lane >> 4;
#pragma unroll
  for (int i = 0; i < 4; ++i)
#pragma unroll
    for (int j = 0; j < 2; ++j) {
      int ncol = n0 + wn + j * 16 + nlane;
#pragma unroll
      for (int r = 0; r < 8; ++r) {
        int mrow = m0 + wm + i * 16 + half * 8 + r;
        float g = cg[i][j][r], u = cu[i][j][r];
        h[(size_t)mrow * FDIM + ncol] = f2bf(u * fast_silu(g));
      }
    }
}

__global__ __launch_bounds__(256) void k_shared_down(
    const unsigned short* __restrict__ h,    // [T][F] bf16
    const float* __restrict__ wd,            // [D][F]  (already [N][K])
    float* __restrict__ out) {               // [T][D]
  __shared__ __align__(16) unsigned short lA[BM * LDK];
  __shared__ __align__(16) unsigned short lB[BN * LDK];
  const int m0 = blockIdx.x * BM, n0 = blockIdx.y * BN;
  const int tid = threadIdx.x, lane = tid & 31, wid = tid >> 5;
  const int wm = (wid >> 2) * 64, wn = (wid & 3) * 32;
  v8f cc[4][2] = {};
  for (int k0 = 0; k0 < FDIM; k0 += BK) {
    __syncthreads();
    for (int task = tid; task < BM * 4; task += 256) {
      int r = task >> 2, v = (task & 3) * 8;
      async_ld_b128(h + (size_t)(m0 + r) * FDIM + k0 + v, lA + r * LDK + v);
    }
    for (int task = tid; task < BN * 8; task += 256) {
      int r = task >> 3, v = (task & 7) * 4;
      const float* wp = wd + (size_t)(n0 + r) * FDIM + k0 + v;
      if ((task & 7) == 0 && k0 + BK < FDIM) __builtin_prefetch(wp + BK, 0, 0);
      float4 w4 = *(const float4*)wp;
      us4 ww; ww.x = f2bf(w4.x); ww.y = f2bf(w4.y); ww.z = f2bf(w4.z); ww.w = f2bf(w4.w);
      *(us4*)(lB + r * LDK + v) = ww;
    }
    wait_async();
    __syncthreads();
    v16bf a[4];
#pragma unroll
    for (int i = 0; i < 4; ++i) a[i] = frag_ld_A(lA + (wm + i * 16) * LDK, lane);
#pragma unroll
    for (int j = 0; j < 2; ++j) {
      v16bf b = frag_ld_B(lB + (wn + j * 16) * LDK, lane);
#pragma unroll
      for (int i = 0; i < 4; ++i) cc[i][j] = WMMA_BF16(a[i], b, cc[i][j]);
    }
  }
  const int nlane = lane & 15, half = lane >> 4;
#pragma unroll
  for (int i = 0; i < 4; ++i)
#pragma unroll
    for (int j = 0; j < 2; ++j) {
      int ncol = n0 + wn + j * 16 + nlane;
#pragma unroll
      for (int r = 0; r < 8; ++r) {
        int mrow = m0 + wm + i * 16 + half * 8 + r;
        out[(size_t)mrow * DIM + ncol] = cc[i][j][r];
      }
    }
}

// ---------------------------------------------------------------- routed path

__global__ __launch_bounds__(256) void k_moe_gateup(
    const unsigned short* __restrict__ xb,   // [T][D] bf16 (unscaled)
    const float* __restrict__ gup,           // [E][D][2F]  ([K][N], needs transpose)
    const int* __restrict__ perm, const float* __restrict__ scl,
    const int* __restrict__ cnt, const int* __restrict__ offs,
    unsigned short* __restrict__ h) {        // [T(sorted)][F] bf16
  const int e = blockIdx.x / MT_PER_E, mt = blockIdx.x % MT_PER_E;
  const int count = cnt[e];
  const int m0 = mt * BM;
  if (m0 >= count) return;
  const int base = offs[e];
  const float* W = gup + (size_t)e * DIM * (2 * FDIM);
  const int n0 = blockIdx.y * BN;            // gate col n0 ; up col n0+FDIM
  __shared__ __align__(16) unsigned short lA[BM * LDK];
  __shared__ __align__(16) unsigned short lG[BN * LDK];
  __shared__ __align__(16) unsigned short lU[BN * LDK];
  __shared__ int   s_tok[BM];
  __shared__ float s_scl[BM];
  const int tid = threadIdx.x, lane = tid & 31, wid = tid >> 5;
  const int wm = (wid >> 2) * 64, wn = (wid & 3) * 32;
  if (tid < BM) {
    int idx = m0 + tid;
    if (idx < count) { int tk = perm[base + idx]; s_tok[tid] = tk; s_scl[tid] = scl[tk]; }
    else             { s_tok[tid] = -1; s_scl[tid] = 0.f; }
  }
  v8f cg[4][2] = {}; v8f cu[4][2] = {};
  for (int k0 = 0; k0 < DIM; k0 += BK) {
    __syncthreads();
    // A: gathered, pre-scaled by sigmoid(router logit)  (needs VALU -> stays sync)
    for (int task = tid; task < BM * 8; task += 256) {
      int r = task >> 3, v = (task & 7) * 4;
      us4 o = {0, 0, 0, 0};
      int tk = s_tok[r];
      if (tk >= 0) {
        us4 d = *(const us4*)(xb + (size_t)tk * DIM + k0 + v);
        float sc = s_scl[r];
        o.x = f2bf(bf2f(d.x) * sc); o.y = f2bf(bf2f(d.y) * sc);
        o.z = f2bf(bf2f(d.z) * sc); o.w = f2bf(bf2f(d.w) * sc);
      }
      *(us4*)(lA + r * LDK + v) = o;
    }
    // B: transpose [K][N] -> LDS [N][K], fp32 -> bf16
    for (int task = tid; task < BK * 32; task += 256) {
      int k = task >> 5, n = (task & 31) * 4;
      const float* rowp = W + (size_t)(k0 + k) * (2 * FDIM);
      if ((task & 31) == 0 && k0 + BK < DIM) {
        __builtin_prefetch(rowp + (size_t)BK * (2 * FDIM) + n0, 0, 0);
        __builtin_prefetch(rowp + (size_t)BK * (2 * FDIM) + FDIM + n0, 0, 0);
      }
      float4 g4 = *(const float4*)(rowp + n0 + n);
      lG[(n + 0) * LDK + k] = f2bf(g4.x); lG[(n + 1) * LDK + k] = f2bf(g4.y);
      lG[(n + 2) * LDK + k] = f2bf(g4.z); lG[(n + 3) * LDK + k] = f2bf(g4.w);
      float4 u4 = *(const float4*)(rowp + FDIM + n0 + n);
      lU[(n + 0) * LDK + k] = f2bf(u4.x); lU[(n + 1) * LDK + k] = f2bf(u4.y);
      lU[(n + 2) * LDK + k] = f2bf(u4.z); lU[(n + 3) * LDK + k] = f2bf(u4.w);
    }
    __syncthreads();
    v16bf a[4];
#pragma unroll
    for (int i = 0; i < 4; ++i) a[i] = frag_ld_A(lA + (wm + i * 16) * LDK, lane);
#pragma unroll
    for (int j = 0; j < 2; ++j) {
      v16bf bg = frag_ld_B(lG + (wn + j * 16) * LDK, lane);
      v16bf bu = frag_ld_B(lU + (wn + j * 16) * LDK, lane);
#pragma unroll
      for (int i = 0; i < 4; ++i) {
        cg[i][j] = WMMA_BF16(a[i], bg, cg[i][j]);
        cu[i][j] = WMMA_BF16(a[i], bu, cu[i][j]);
      }
    }
  }
  const int nlane = lane & 15, half = lane >> 4;
#pragma unroll
  for (int i = 0; i < 4; ++i)
#pragma unroll
    for (int j = 0; j < 2; ++j) {
      int ncol = n0 + wn + j * 16 + nlane;
#pragma unroll
      for (int r = 0; r < 8; ++r) {
        int ml = wm + i * 16 + half * 8 + r;
        if (m0 + ml < count) {
          float g = cg[i][j][r], u = cu[i][j][r];
          h[(size_t)(base + m0 + ml) * FDIM + ncol] = f2bf(u * fast_silu(g));
        }
      }
    }
}

__global__ __launch_bounds__(256) void k_moe_down(
    const unsigned short* __restrict__ h,    // [T(sorted)][F] bf16
    const float* __restrict__ dwn,           // [E][F][D]  ([K][N], needs transpose)
    const int* __restrict__ perm,
    const int* __restrict__ cnt, const int* __restrict__ offs,
    float* __restrict__ out) {               // [T][D] (+=)
  const int e = blockIdx.x / MT_PER_E, mt = blockIdx.x % MT_PER_E;
  const int count = cnt[e];
  const int m0 = mt * BM;
  if (m0 >= count) return;
  const int base = offs[e];
  const float* W = dwn + (size_t)e * FDIM * DIM;
  const int n0 = blockIdx.y * BN;
  __shared__ __align__(16) unsigned short lA[BM * LDK];
  __shared__ __align__(16) unsigned short lB[BN * LDK];
  __shared__ int s_tok[BM];
  const int tid = threadIdx.x, lane = tid & 31, wid = tid >> 5;
  const int wm = (wid >> 2) * 64, wn = (wid & 3) * 32;
  if (tid < BM) {
    int idx = m0 + tid;
    s_tok[tid] = (idx < count) ? perm[base + idx] : 0;
  }
  v8f cc[4][2] = {};
  for (int k0 = 0; k0 < FDIM; k0 += BK) {
    __syncthreads();
    for (int task = tid; task < BM * 4; task += 256) {
      int r = task >> 2, v = (task & 3) * 8;
      unsigned short* dst = lA + r * LDK + v;
      if (m0 + r < count) {
        async_ld_b128(h + (size_t)(base + m0 + r) * FDIM + k0 + v, dst);
      } else {
        us8 z = {0, 0, 0, 0, 0, 0, 0, 0};
        *(us8*)dst = z;
      }
    }
    for (int task = tid; task < BK * 32; task += 256) {
      int k = task >> 5, n = (task & 31) * 4;
      const float* rowp = W + (size_t)(k0 + k) * DIM;
      if ((task & 31) == 0 && k0 + BK < FDIM)
        __builtin_prefetch(rowp + (size_t)BK * DIM + n0, 0, 0);
      float4 w4 = *(const float4*)(rowp + n0 + n);
      lB[(n + 0) * LDK + k] = f2bf(w4.x); lB[(n + 1) * LDK + k] = f2bf(w4.y);
      lB[(n + 2) * LDK + k] = f2bf(w4.z); lB[(n + 3) * LDK + k] = f2bf(w4.w);
    }
    wait_async();
    __syncthreads();
    v16bf a[4];
#pragma unroll
    for (int i = 0; i < 4; ++i) a[i] = frag_ld_A(lA + (wm + i * 16) * LDK, lane);
#pragma unroll
    for (int j = 0; j < 2; ++j) {
      v16bf b = frag_ld_B(lB + (wn + j * 16) * LDK, lane);
#pragma unroll
      for (int i = 0; i < 4; ++i) cc[i][j] = WMMA_BF16(a[i], b, cc[i][j]);
    }
  }
  const int nlane = lane & 15, half = lane >> 4;
#pragma unroll
  for (int i = 0; i < 4; ++i)
#pragma unroll
    for (int j = 0; j < 2; ++j) {
      int ncol = n0 + wn + j * 16 + nlane;
#pragma unroll
      for (int r = 0; r < 8; ++r) {
        int ml = wm + i * 16 + half * 8 + r;
        if (m0 + ml < count) {
          out[(size_t)s_tok[ml] * DIM + ncol] += cc[i][j][r];  // race-free: buckets partition tokens
        }
      }
    }
}

// ---------------------------------------------------------------- launch

extern "C" void kernel_launch(void* const* d_in, const int* in_sizes, int n_in,
                              void* d_out, int out_size, void* d_ws, size_t ws_size,
                              hipStream_t stream) {
  const float* x   = (const float*)d_in[0];   // [2,2048,1024]
  const float* rw  = (const float*)d_in[1];   // [8,1024]
  const float* gup = (const float*)d_in[2];   // [8,1024,4096]
  const float* dwn = (const float*)d_in[3];   // [8,2048,1024]
  const float* shg = (const float*)d_in[4];   // [2048,1024]
  const float* shu = (const float*)d_in[5];   // [2048,1024]
  const float* shd = (const float*)d_in[6];   // [1024,2048]
  float* out = (float*)d_out;                 // [4096,1024]

  char* w = (char*)d_ws;
  unsigned short* xb   = (unsigned short*)w; w += (size_t)T_TOK * DIM * 2;   // 8 MiB
  unsigned short* hbuf = (unsigned short*)w; w += (size_t)T_TOK * FDIM * 2;  // 16 MiB (reused)
  int*   tope = (int*)w;   w += T_TOK * 4;
  float* scl  = (float*)w; w += T_TOK * 4;
  int*   perm = (int*)w;   w += T_TOK * 4;
  int*   cnt  = (int*)w;   w += 64;
  int*   offs = (int*)w;   w += 64;
  int*   curs = (int*)w;   w += 64;

  (void)hipMemsetAsync(cnt, 0, NEXP * sizeof(int), stream);

  k_convert_x<<<(T_TOK * DIM) / (256 * 4), 256, 0, stream>>>(x, xb);
  k_router<<<T_TOK / 8, 256, 0, stream>>>(x, rw, tope, scl, cnt);
  k_scan8<<<1, 32, 0, stream>>>(cnt, offs, curs);
  k_assign<<<T_TOK / 256, 256, 0, stream>>>(tope, curs, perm);

  // shared-expert MLP (writes all of out)
  k_shared_gateup<<<dim3(T_TOK / BM, FDIM / BN), 256, 0, stream>>>(xb, shg, shu, hbuf);
  k_shared_down<<<dim3(T_TOK / BM, DIM / BN), 256, 0, stream>>>(hbuf, shd, out);

  // routed top-1 expert MLP (accumulates into out); hbuf reused
  k_moe_gateup<<<dim3(NEXP * MT_PER_E, FDIM / BN), 256, 0, stream>>>(
      xb, gup, perm, scl, cnt, offs, hbuf);
  k_moe_down<<<dim3(NEXP * MT_PER_E, DIM / BN), 256, 0, stream>>>(
      hbuf, dwn, perm, cnt, offs, out);
}
